// GNNStack_86466281603776
// MI455X (gfx1250) — matrix operand order repaired and checked by reading, not compiled, not run on hardware
//
#include <hip/hip_runtime.h>

#define N_NODES 50000
#define N_EDGES 800000
#define IN_DIM  128
#define HID     32
#define OUT_DIM 40
#define OUT_PAD 48   // W2 padded to 48 rows (3 full 16-col N-tiles)

typedef __attribute__((ext_vector_type(16))) _Float16 v16h;
typedef __attribute__((ext_vector_type(8)))  float    v8f;

// Probe-verified CDNA5 builtin: D(f32 16x16) = A(f16 16x32) * B(f16 32x16) + C
static __device__ inline v8f wmma(v16h a, v16h b, v8f c) {
    return __builtin_amdgcn_wmma_f32_16x16x32_f16(
        /*neg_a=*/false, a, /*neg_b=*/false, b,
        /*c_mod=*/(short)0, c, /*reuse_a=*/false, /*reuse_b=*/false);
}

// f16 weight scratch sizes (elements)
#define WL_H_ELEMS  (HID * IN_DIM)       // 4096
#define WR_H_ELEMS  (HID * IN_DIM)       // 4096
#define W1_H_ELEMS  (HID * HID)          // 1024
#define W2_H_ELEMS  (OUT_PAD * HID)      // 1536 (rows 40..47 zero)
#define WTOT_H_ELEMS (WL_H_ELEMS + WR_H_ELEMS + W1_H_ELEMS + W2_H_ELEMS)

// ---------------------------------------------------------------- zero ws
__global__ __launch_bounds__(256) void zero_ws_kernel(float* __restrict__ p, int n) {
    int stride = gridDim.x * blockDim.x;
    for (int i = blockIdx.x * blockDim.x + threadIdx.x; i < n; i += stride)
        p[i] = 0.0f;
}

// -------------------------------------- one-shot f32 -> f16 weight repack
__global__ __launch_bounds__(256) void pack_weights_kernel(
    const float* __restrict__ Wl, const float* __restrict__ Wr,
    const float* __restrict__ W1, const float* __restrict__ W2,
    _Float16* __restrict__ wh) {
    int i = blockIdx.x * blockDim.x + threadIdx.x;
    if (i >= WTOT_H_ELEMS) return;
    float v;
    if (i < WL_H_ELEMS) {
        v = Wl[i];
    } else if (i < WL_H_ELEMS + WR_H_ELEMS) {
        v = Wr[i - WL_H_ELEMS];
    } else if (i < WL_H_ELEMS + WR_H_ELEMS + W1_H_ELEMS) {
        v = W1[i - WL_H_ELEMS - WR_H_ELEMS];
    } else {
        int k   = i - WL_H_ELEMS - WR_H_ELEMS - W1_H_ELEMS;
        int row = k / HID;
        v = (row < OUT_DIM) ? W2[k] : 0.0f;   // zero-padded rows 40..47
    }
    wh[i] = (_Float16)v;
}

// ------------------------------------------------- scatter-add (one wave/edge)
__global__ __launch_bounds__(256) void scatter_kernel(
    const float* __restrict__ x, const int* __restrict__ ei,
    float* __restrict__ agg, float* __restrict__ cnt) {
    int edge = blockIdx.x * 8 + (threadIdx.x >> 5);   // 8 waves per block
    int lane = threadIdx.x & 31;
    if (edge >= N_EDGES) return;
    int src = ei[edge];
    int dst = ei[N_EDGES + edge];
    // 32 lanes x float4 = full 128-dim row
    const float4 v = *(const float4*)(x + (size_t)src * IN_DIM + lane * 4);
    float* d = agg + (size_t)dst * IN_DIM + lane * 4;
    atomicAdd(d + 0, v.x);
    atomicAdd(d + 1, v.y);
    atomicAdd(d + 2, v.z);
    atomicAdd(d + 3, v.w);
    if (lane == 0) atomicAdd(cnt + dst, 1.0f);
}

// ---------------------------------- fused per-16-node-tile transform (1 wave)
__global__ __launch_bounds__(32) void node_transform_kernel(
    const float* __restrict__ x,   const float* __restrict__ agg,
    const float* __restrict__ cnt, const _Float16* __restrict__ wh,
    const float* __restrict__ b1,  const float* __restrict__ b2,
    float* __restrict__ out) {
    __shared__ float lds_h[16][HID + 1];
    __shared__ float lds_o[16][48];

    const int lane = threadIdx.x;      // 0..31
    const int m    = lane & 15;        // row within tile (A layout), column (B/C layouts)
    const int g    = lane >> 4;        // half-wave group
    const int tile = blockIdx.x;
    const int node = tile * 16 + m;    // 50000 = 3125*16: every tile is full

    const _Float16* Wlh = wh;
    const _Float16* Wrh = wh + WL_H_ELEMS;
    const _Float16* W1h = wh + WL_H_ELEMS + WR_H_ELEMS;
    const _Float16* W2h = wh + WL_H_ELEMS + WR_H_ELEMS + W1_H_ELEMS;

    // 1-ulp rcp: result feeds an f16 convert, so approximation is lossless here
    const float rc  = __builtin_amdgcn_rcpf(fmaxf(cnt[node], 1.0f));
    const float* xr = x   + (size_t)node * IN_DIM;
    const float* ar = agg + (size_t)node * IN_DIM;

    // ---- layer 1: h = x @ Wl^T + mean @ Wr^T   (K=128 -> 4 chunks of 32)
    v8f c0 = {}, c1 = {};
#pragma unroll
    for (int kc = 0; kc < IN_DIM / 32; ++kc) {
        const int kb = kc * 32;
        v16h ax, am;
#pragma unroll
        for (int j = 0; j < 8; ++j) {             // contiguous k = 16*g + s (matches B)
            ax[j]     = (_Float16)xr[kb + 16 * g + j];
            ax[8 + j] = (_Float16)xr[kb + 16 * g + 8 + j];
            am[j]     = (_Float16)(ar[kb + 16 * g + j] * rc);
            am[8 + j] = (_Float16)(ar[kb + 16 * g + 8 + j] * rc);
        }
#pragma unroll
        for (int nt = 0; nt < 2; ++nt) {          // HID=32 -> 2 N-tiles
            const int n = nt * 16 + m;            // B[k][n] = W[n][k]
            // 16 contiguous f16 = one 32B-aligned vector load (2x b128)
            const v16h bl = *(const v16h*)(Wlh + n * IN_DIM + kb + 16 * g);
            const v16h br = *(const v16h*)(Wrh + n * IN_DIM + kb + 16 * g);
            if (nt == 0) { c0 = wmma(ax, bl, c0); c0 = wmma(am, br, c0); }
            else         { c1 = wmma(ax, bl, c1); c1 = wmma(am, br, c1); }
        }
    }

    // ---- row L2-normalize + relu directly on C-layout fragments
    float sq[8];
#pragma unroll
    for (int r = 0; r < 8; ++r) sq[r] = c0[r] * c0[r] + c1[r] * c1[r];
#pragma unroll
    for (int off = 1; off < 16; off <<= 1) {      // reduce inside 16-lane halves
#pragma unroll
        for (int r = 0; r < 8; ++r) sq[r] += __shfl_xor(sq[r], off, 32);
    }
#pragma unroll
    for (int r = 0; r < 8; ++r) {
        const float s = 1.0f / fmaxf(sqrtf(sq[r]), 1e-12f);
        c0[r] = fmaxf(c0[r] * s, 0.0f);
        c1[r] = fmaxf(c1[r] * s, 0.0f);
    }

    // ---- stage h via LDS: C layout -> A layout
#pragma unroll
    for (int r = 0; r < 8; ++r) {
        lds_h[r + 8 * g][m]      = c0[r];
        lds_h[r + 8 * g][16 + m] = c1[r];
    }
    __syncthreads();
    v16h ah;
#pragma unroll
    for (int j = 0; j < 8; ++j) {
        ah[j]     = (_Float16)lds_h[m][16 * g + j];
        ah[8 + j] = (_Float16)lds_h[m][16 * g + 8 + j];
    }
    __syncthreads();

    // ---- layer 2: h2 = h @ W1^T + b1   (bias preloaded into accumulators)
    v8f c2a, c2b;
    const float bb0 = b1[m], bb1 = b1[16 + m];
#pragma unroll
    for (int r = 0; r < 8; ++r) { c2a[r] = bb0; c2b[r] = bb1; }
    {
        const v16h bw0 = *(const v16h*)(W1h + m * HID + 16 * g);
        const v16h bw1 = *(const v16h*)(W1h + (16 + m) * HID + 16 * g);
        c2a = wmma(ah, bw0, c2a);
        c2b = wmma(ah, bw1, c2b);
    }

    // ---- stage h2 via LDS
#pragma unroll
    for (int r = 0; r < 8; ++r) {
        lds_h[r + 8 * g][m]      = c2a[r];
        lds_h[r + 8 * g][16 + m] = c2b[r];
    }
    __syncthreads();
    v16h ah2;
#pragma unroll
    for (int j = 0; j < 8; ++j) {
        ah2[j]     = (_Float16)lds_h[m][16 * g + j];
        ah2[8 + j] = (_Float16)lds_h[m][16 * g + 8 + j];
    }

    // ---- layer 3: h3 = h2 @ W2^T + b2  (W2 pre-padded to 48 rows: no guards)
    v8f c3[3];
#pragma unroll
    for (int nt = 0; nt < 3; ++nt) {
        const int n = nt * 16 + m;
        const float bb = (n < OUT_DIM) ? b2[n] : 0.0f;
#pragma unroll
        for (int r = 0; r < 8; ++r) c3[nt][r] = bb;
        const v16h bw = *(const v16h*)(W2h + n * HID + 16 * g);
        c3[nt] = wmma(ah2, bw, c3[nt]);
    }

    // ---- log-softmax over 40 classes per row
#pragma unroll
    for (int nt = 0; nt < 3; ++nt)
#pragma unroll
        for (int r = 0; r < 8; ++r)
            lds_o[r + 8 * g][nt * 16 + m] = c3[nt][r];
    __syncthreads();
    if (lane < 16) {
        float mx = -3.0e38f;
        for (int c = 0; c < OUT_DIM; ++c) mx = fmaxf(mx, lds_o[lane][c]);
        float s = 0.0f;
        for (int c = 0; c < OUT_DIM; ++c) s += expf(lds_o[lane][c] - mx);
        const float lse = mx + logf(s);
        float* orow = out + (size_t)(tile * 16 + lane) * OUT_DIM;
        for (int c = 0; c < OUT_DIM; ++c) orow[c] = lds_o[lane][c] - lse;
    }
}

// ------------------------------------------------------------------ launch
extern "C" void kernel_launch(void* const* d_in, const int* in_sizes, int n_in,
                              void* d_out, int out_size, void* d_ws, size_t ws_size,
                              hipStream_t stream) {
    (void)in_sizes; (void)n_in; (void)out_size; (void)ws_size;
    const float* x  = (const float*)d_in[0];
    const int*   ei = (const int*)  d_in[1];
    const float* Wl = (const float*)d_in[2];
    const float* Wr = (const float*)d_in[3];
    const float* W1 = (const float*)d_in[4];
    const float* b1 = (const float*)d_in[5];
    const float* W2 = (const float*)d_in[6];
    const float* b2 = (const float*)d_in[7];
    float* out = (float*)d_out;

    float*    agg = (float*)d_ws;                      // [N, 128] f32
    float*    cnt = agg + (size_t)N_NODES * IN_DIM;    // [N] f32
    _Float16* wh  = (_Float16*)(cnt + N_NODES);        // packed f16 weights (32B-aligned)

    zero_ws_kernel<<<2048, 256, 0, stream>>>(agg, N_NODES * IN_DIM + N_NODES);
    pack_weights_kernel<<<(WTOT_H_ELEMS + 255) / 256, 256, 0, stream>>>(
        Wl, Wr, W1, W2, wh);
    scatter_kernel<<<(N_EDGES + 7) / 8, 256, 0, stream>>>(x, ei, agg, cnt);
    node_transform_kernel<<<N_NODES / 16, 32, 0, stream>>>(
        x, agg, cnt, wh, b1, b2, out);
}
